// Transformer_22153441313324
// MI455X (gfx1250) — matrix operand order repaired and checked by reading, not compiled, not run on hardware
//
#include <hip/hip_runtime.h>

// ---------------------------------------------------------------------------
// Transformer forward for MI455X (gfx1250). All matmuls (QKV proj, QK^T, A*V,
// WO, FF, vocab proj) run through v_wmma_f32_16x16x32_bf16 with fp32
// accumulation; activations/weights converted fp32->bf16 while staging to LDS.
// Software-pipelined: global b128 loads for tile kt+1 are issued before the
// WMMAs of tile kt, so HBM latency is hidden behind compute + LDS traffic.
// All GEMM shapes issued by the launcher satisfy M%64==0, N%64==0, K%32==0,
// so the kernel carries no bounds checks (straight-line staging).
// ---------------------------------------------------------------------------

typedef __attribute__((ext_vector_type(16))) __bf16 v16bf;
typedef __attribute__((ext_vector_type(8)))  __bf16 v8bf;
typedef __attribute__((ext_vector_type(4)))  __bf16 v4bf;
typedef __attribute__((ext_vector_type(8)))  float  v8f;

#define TILE_M 64
#define TILE_N 64
#define TILE_K 32
#define LDS_STRIDE 40  // bf16 elems per row: 80 bytes, 16B-aligned rows

// --- WMMA fragment loads (layouts per CDNA5 ISA 7.12.2, wave32) -------------
// A 16x32 bf16: lane l holds row (l&15); elems 0..7 = K[(l>>4)*8 .. +7],
//               elems 8..15 = K[16+(l>>4)*8 .. +7].
static __device__ inline v16bf load_frag_a(const __bf16 (*T)[LDS_STRIDE],
                                           int row, int hi) {
  const v8bf x0 = *(const v8bf*)&T[row][hi * 8];
  const v8bf x1 = *(const v8bf*)&T[row][16 + hi * 8];
  v16bf r;
#pragma unroll
  for (int i = 0; i < 8; ++i) { r[i] = x0[i]; r[8 + i] = x1[i]; }
  return r;
}
// B 32x16 bf16: lane l holds col (l&15); elems 0..7 = K[(l>>4)*8 .. +7],
//               elems 8..15 = K[16+(l>>4)*8 .. +7].
// LDS tile is stored transposed: T[n][k].
static __device__ inline v16bf load_frag_b(const __bf16 (*T)[LDS_STRIDE],
                                           int col, int hi) {
  const v8bf x0 = *(const v8bf*)&T[col][hi * 8];
  const v8bf x1 = *(const v8bf*)&T[col][16 + hi * 8];
  v16bf r;
#pragma unroll
  for (int i = 0; i < 8; ++i) { r[i] = x0[i]; r[8 + i] = x1[i]; }
  return r;
}

// Generalized batched GEMM: C[z] = alpha * A[z] * B[z] + bias.
//   z decomposition: zo = z / zdiv, zi = z % zdiv; tensor base += zo*s1 + zi*s2
//   A element (m,k)  at  m*lda + k
//   B element (k,n):
//     BMODE==1 (contig-n): (n/nbb)*bsb + k*bsk + (n%nbb)   [nbb%4==0, bsk%4==0]
//     BMODE==2 (contig-k): n*bsb + k                        [QK^T slices]
//   C element (m,n)  at  m*ldc + n
template <int BMODE>
__global__ __launch_bounds__(128)
void gemm_wmma_bf16(const float* __restrict__ A, long long az1, long long az2, int lda,
                    const float* __restrict__ Bp, long long bz1, long long bz2,
                    int nbb, long long bsb, int bsk,
                    const float* __restrict__ bias,
                    float* __restrict__ C, long long cz1, long long cz2, int ldc,
                    int M, int N, int K, int zdiv, float alpha) {
  __shared__ __attribute__((aligned(16))) __bf16 As[TILE_M][LDS_STRIDE];
  __shared__ __attribute__((aligned(16))) __bf16 Bs[TILE_N][LDS_STRIDE]; // [n][k]

  const int tid  = threadIdx.x;
  const int lane = tid & 31;
  const int wave = tid >> 5;            // 0..3
  const int wm   = (wave >> 1) * 32;    // wave's row offset in 64x64 tile
  const int wn   = (wave & 1) * 32;     // wave's col offset
  const int lo   = lane & 15;
  const int hi   = lane >> 4;

  const int z  = blockIdx.z;
  const int zo = z / zdiv, zi = z % zdiv;
  A  += zo * az1 + zi * az2;
  Bp += zo * bz1 + zi * bz2;
  C  += zo * cz1 + zi * cz2;

  const int m0 = blockIdx.y * TILE_M;
  const int n0 = blockIdx.x * TILE_N;

  // Per-thread staging coordinates (fixed across K-steps).
  const int ar  = tid >> 1;                   // A row 0..63
  const int ac4 = (tid & 1) * 16;             // A col base (+j*4)
  const float* Arow = A + (long long)(m0 + ar) * lda + ac4;

  // B per-j column bases (independent of kt).
  const float* bbase[4];
  int bk[4];   // BMODE1: k-row within tile ; BMODE2: k4 col offset
  int bn[4];   // BMODE1: n4 base           ; BMODE2: n row
#pragma unroll
  for (int j = 0; j < 4; ++j) {
    int q = j * 128 + tid;
    if (BMODE == 1) {
      bk[j] = q >> 4;                       // 0..31
      bn[j] = (q & 15) * 4;                 // 0..60
      int gn = n0 + bn[j];
      bbase[j] = Bp + (long long)(gn / nbb) * bsb + (gn % nbb);
    } else {
      bn[j] = q >> 3;                       // 0..63
      bk[j] = (q & 7) * 4;                  // 0..28
      bbase[j] = Bp + (long long)(n0 + bn[j]) * bsb + bk[j];
    }
  }

  v8f acc00 = {}, acc01 = {}, acc10 = {}, acc11 = {};

  // -------- prologue: fetch tile kt=0 into registers --------
  float4 ra[4], rb[4];
#pragma unroll
  for (int j = 0; j < 4; ++j) ra[j] = *(const float4*)(Arow + j * 4);
#pragma unroll
  for (int j = 0; j < 4; ++j)
    rb[j] = (BMODE == 1)
                ? *(const float4*)(bbase[j] + (long long)bk[j] * bsk)
                : *(const float4*)(bbase[j]);

  for (int kt = 0; kt < K; kt += TILE_K) {
    // ---- convert + store current registers to LDS ----
#pragma unroll
    for (int j = 0; j < 4; ++j) {
      v4bf h;
      h[0] = (__bf16)ra[j].x; h[1] = (__bf16)ra[j].y;
      h[2] = (__bf16)ra[j].z; h[3] = (__bf16)ra[j].w;
      *(v4bf*)&As[ar][ac4 + j * 4] = h;
    }
    if (BMODE == 1) {
#pragma unroll
      for (int j = 0; j < 4; ++j) {
        Bs[bn[j] + 0][bk[j]] = (__bf16)rb[j].x;
        Bs[bn[j] + 1][bk[j]] = (__bf16)rb[j].y;
        Bs[bn[j] + 2][bk[j]] = (__bf16)rb[j].z;
        Bs[bn[j] + 3][bk[j]] = (__bf16)rb[j].w;
      }
    } else {
#pragma unroll
      for (int j = 0; j < 4; ++j) {
        v4bf h;
        h[0] = (__bf16)rb[j].x; h[1] = (__bf16)rb[j].y;
        h[2] = (__bf16)rb[j].z; h[3] = (__bf16)rb[j].w;
        *(v4bf*)&Bs[bn[j]][bk[j]] = h;
      }
    }
    __syncthreads();

    // ---- issue global loads for tile kt+1 (waited next iteration) ----
    const int kn = kt + TILE_K;
    if (kn < K) {
#pragma unroll
      for (int j = 0; j < 4; ++j) ra[j] = *(const float4*)(Arow + kn + j * 4);
#pragma unroll
      for (int j = 0; j < 4; ++j)
        rb[j] = (BMODE == 1)
                    ? *(const float4*)(bbase[j] + (long long)(kn + bk[j]) * bsk)
                    : *(const float4*)(bbase[j] + kn);
    }

    // ---- WMMA on tile kt ----
    v16bf a0 = load_frag_a(As, wm + lo,      hi);
    v16bf a1 = load_frag_a(As, wm + 16 + lo, hi);
    v16bf b0 = load_frag_b(Bs, wn + lo,      hi);
    v16bf b1 = load_frag_b(Bs, wn + 16 + lo, hi);

    acc00 = __builtin_amdgcn_wmma_f32_16x16x32_bf16(false, a0, false, b0,
                                                    (short)0, acc00, false, false);
    acc01 = __builtin_amdgcn_wmma_f32_16x16x32_bf16(false, a0, false, b1,
                                                    (short)0, acc01, false, false);
    acc10 = __builtin_amdgcn_wmma_f32_16x16x32_bf16(false, a1, false, b0,
                                                    (short)0, acc10, false, false);
    acc11 = __builtin_amdgcn_wmma_f32_16x16x32_bf16(false, a1, false, b1,
                                                    (short)0, acc11, false, false);
    __syncthreads();
  }

  // Store: C/D layout — lane group selects M half (hi*8 + vgpr r), N = lo.
  const int gnA = n0 + wn + lo;
  const int gnB = gnA + 16;
  const float bA = bias ? bias[gnA] : 0.0f;
  const float bB = bias ? bias[gnB] : 0.0f;
#pragma unroll
  for (int r = 0; r < 8; ++r) {
    int gmA = m0 + wm + hi * 8 + r;
    int gmB = gmA + 16;
    C[(long long)gmA * ldc + gnA] = acc00[r] * alpha + bA;
    C[(long long)gmA * ldc + gnB] = acc01[r] * alpha + bB;
    C[(long long)gmB * ldc + gnA] = acc10[r] * alpha + bA;
    C[(long long)gmB * ldc + gnB] = acc11[r] * alpha + bB;
  }
}

// --- embedding + positional encoding ---------------------------------------
__global__ void embed_kernel(const int* __restrict__ src,
                             const float* __restrict__ emb,
                             const float* __restrict__ pos,
                             float* __restrict__ out, int S, int D) {
  long long r = blockIdx.x;           // row = b*S + s
  int s = (int)(r % S);
  int tok = src[r];
  const float* e = emb + (long long)tok * D;
  const float* p = pos + (long long)s * D;
  float* o = out + r * D;
  for (int d = threadIdx.x; d < D; d += blockDim.x) o[d] = e[d] + p[d];
}

// --- LayerNorm over last dim (D=512) with fused residual + optional mask ---
__global__ __launch_bounds__(256)
void layernorm_kernel(const float* __restrict__ x, const float* __restrict__ res,
                      const float* __restrict__ gamma, const float* __restrict__ beta,
                      const int* __restrict__ srcmask,
                      float* __restrict__ out, int D) {
  __shared__ float row[512];
  __shared__ float red[256];
  long long r = blockIdx.x;
  int tid = threadIdx.x;
  const float* xr = x + r * D;
  const float* rr = res ? res + r * D : nullptr;
  for (int d = tid; d < D; d += 256) row[d] = xr[d] + (rr ? rr[d] : 0.0f);
  __syncthreads();

  float s = 0.0f;
  for (int d = tid; d < D; d += 256) s += row[d];
  red[tid] = s; __syncthreads();
  for (int off = 128; off > 0; off >>= 1) {
    if (tid < off) red[tid] += red[tid + off];
    __syncthreads();
  }
  float mean = red[0] / (float)D;
  __syncthreads();

  float v = 0.0f;
  for (int d = tid; d < D; d += 256) { float t = row[d] - mean; v += t * t; }
  red[tid] = v; __syncthreads();
  for (int off = 128; off > 0; off >>= 1) {
    if (tid < off) red[tid] += red[tid + off];
    __syncthreads();
  }
  float var = red[0] / (float)D;
  float rstd = rsqrtf(var + 1e-3f);
  float mk = srcmask ? ((srcmask[r] != 0) ? 1.0f : 0.0f) : 1.0f;
  float* o = out + r * D;
  for (int d = tid; d < D; d += 256)
    o[d] = ((row[d] - mean) * rstd * gamma[d] + beta[d]) * mk;
}

// --- row softmax with POST-softmax multiplicative causal mask ---------------
__global__ __launch_bounds__(256)
void softmax_kernel(float* __restrict__ sc, int S, int causal) {
  __shared__ float red[256];
  int s = blockIdx.x, z = blockIdx.y, tid = threadIdx.x;
  float* row = sc + ((long long)z * S + s) * S;

  float m = -3.0e38f;
  for (int t = tid; t < S; t += 256) m = fmaxf(m, row[t]);
  red[tid] = m; __syncthreads();
  for (int off = 128; off > 0; off >>= 1) {
    if (tid < off) red[tid] = fmaxf(red[tid], red[tid + off]);
    __syncthreads();
  }
  m = red[0]; __syncthreads();

  float sum = 0.0f;
  for (int t = tid; t < S; t += 256) {
    float e = __expf(row[t] - m);
    row[t] = e;
    sum += e;
  }
  red[tid] = sum; __syncthreads();
  for (int off = 128; off > 0; off >>= 1) {
    if (tid < off) red[tid] += red[tid + off];
    __syncthreads();
  }
  float inv = 1.0f / red[0];
  for (int t = tid; t < S; t += 256) {
    float p = row[t] * inv;
    if (causal && t > s) p = 0.0f;  // reference masks AFTER softmax
    row[t] = p;
  }
}

// --- feature concat [r,0:D]=a, [r,D:2D]=b ----------------------------------
__global__ void concat_kernel(const float* __restrict__ a, const float* __restrict__ b,
                              float* __restrict__ out, int D) {
  long long r = blockIdx.x;
  for (int d = threadIdx.x; d < 2 * D; d += blockDim.x)
    out[r * 2 * D + d] = (d < D) ? a[r * D + d] : b[r * D + (d - D)];
}

// ---------------------------------------------------------------------------
extern "C" void kernel_launch(void* const* d_in, const int* in_sizes, int n_in,
                              void* d_out, int out_size, void* d_ws, size_t ws_size,
                              hipStream_t stream) {
  constexpr int B = 4, S = 512, D = 512, H = 8, HD = 64;
  constexpr int LE = 4, LD_ = 4, VT = 32000;
  constexpr long long ROWS = (long long)B * S;            // 2048
  constexpr long long WQKV_E = (long long)H * D * HD;     // per layer (din=512)
  constexpr long long WQKV_X = (long long)H * 2 * D * HD; // per layer (din=1024)
  constexpr long long WSQ = (long long)D * D;

  // Inputs (flattened setup_inputs dict order)
  const int*   src     = (const int*)d_in[0];
  /* d_in[1] = target: unused by reference forward */
  const float* pos     = (const float*)d_in[2];
  const float* emb_src = (const float*)d_in[3];
  const float* emb_tgt = (const float*)d_in[4];
  const float* eWQ = (const float*)d_in[5],  *eBQ = (const float*)d_in[6];
  const float* eWK = (const float*)d_in[7],  *eBK = (const float*)d_in[8];
  const float* eWV = (const float*)d_in[9],  *eBV = (const float*)d_in[10];
  const float* eWO = (const float*)d_in[11], *eBO = (const float*)d_in[12];
  const float* eWff = (const float*)d_in[13], *eBff = (const float*)d_in[14];
  const float* eG  = (const float*)d_in[15], *eB  = (const float*)d_in[16];
  const float* mWQ = (const float*)d_in[17], *mBQ = (const float*)d_in[18];
  const float* mWK = (const float*)d_in[19], *mBK = (const float*)d_in[20];
  const float* mWV = (const float*)d_in[21], *mBV = (const float*)d_in[22];
  const float* mWO = (const float*)d_in[23], *mBO = (const float*)d_in[24];
  const float* xWQ = (const float*)d_in[25], *xBQ = (const float*)d_in[26];
  const float* xWK = (const float*)d_in[27], *xBK = (const float*)d_in[28];
  const float* xWV = (const float*)d_in[29], *xBV = (const float*)d_in[30];
  const float* xWO = (const float*)d_in[31], *xBO = (const float*)d_in[32];
  const float* dWff = (const float*)d_in[33], *dBff = (const float*)d_in[34];
  const float* dG  = (const float*)d_in[35], *dB  = (const float*)d_in[36];
  const float* Wout = (const float*)d_in[37], *bout = (const float*)d_in[38];

  // Workspace layout (floats)
  float* ws = (float*)d_ws;
  long long o = 0;
  float* buf_x   = ws + o; o += ROWS * D;
  float* buf_enc = ws + o; o += (long long)LE * ROWS * D;
  float* buf_q   = ws + o; o += ROWS * D;
  float* buf_k   = ws + o; o += ROWS * D;
  float* buf_v   = ws + o; o += ROWS * D;
  float* buf_sc  = ws + o; o += (long long)B * H * S * S;
  float* buf_o   = ws + o; o += ROWS * D;
  float* buf_mha = ws + o; o += ROWS * D;
  float* buf_t   = ws + o; o += ROWS * D;
  float* buf_t2  = ws + o; o += ROWS * D;
  float* buf_ff  = ws + o; o += ROWS * D;
  float* buf_cat = ws + o; o += ROWS * 2 * D;

  auto gemm = [&](const float* A, long long az1, long long az2, int lda,
                  const float* Bp, long long bz1, long long bz2,
                  int nbb, long long bsb, int bsk,
                  const float* bias, float* C, long long cz1, long long cz2, int ldc,
                  int M, int N, int K, int Z, int zdiv, float alpha) {
    dim3 g((N + TILE_N - 1) / TILE_N, (M + TILE_M - 1) / TILE_M, Z);
    if (nbb == 1)  // transposed slice (QK^T): B(k,n) = n*bsb + k
      gemm_wmma_bf16<2><<<g, 128, 0, stream>>>(A, az1, az2, lda, Bp, bz1, bz2, nbb,
                                               bsb, bsk, bias, C, cz1, cz2, ldc,
                                               M, N, K, zdiv, alpha);
    else
      gemm_wmma_bf16<1><<<g, 128, 0, stream>>>(A, az1, az2, lda, Bp, bz1, bz2, nbb,
                                               bsb, bsk, bias, C, cz1, cz2, ldc,
                                               M, N, K, zdiv, alpha);
  };

  // Multi-head attention: x[ROWS,din] -> outp[ROWS,512]
  auto mha = [&](const float* xin, int din,
                 const float* WQ, const float* bQ, const float* WK, const float* bK,
                 const float* WV, const float* bV, const float* WO, const float* bO,
                 int causal, float* outp) {
    gemm(xin, 0, 0, din, WQ, 0, 0, HD, (long long)din * HD, HD, bQ,
         buf_q, 0, 0, D, (int)ROWS, D, din, 1, 1, 1.0f);
    gemm(xin, 0, 0, din, WK, 0, 0, HD, (long long)din * HD, HD, bK,
         buf_k, 0, 0, D, (int)ROWS, D, din, 1, 1, 1.0f);
    gemm(xin, 0, 0, din, WV, 0, 0, HD, (long long)din * HD, HD, bV,
         buf_v, 0, 0, D, (int)ROWS, D, din, 1, 1, 1.0f);
    // scores[z=b*H+h] = Q K^T / 8
    gemm(buf_q, (long long)S * D, HD, D,
         buf_k, (long long)S * D, HD, 1, (long long)D, 1,
         nullptr, buf_sc, (long long)H * S * S, (long long)S * S, S,
         S, S, HD, B * H, H, 0.125f);
    softmax_kernel<<<dim3(S, B * H), 256, 0, stream>>>(buf_sc, S, causal);
    // o[z] = A V   (V slice: nbb=64, bsk=512)
    gemm(buf_sc, (long long)H * S * S, (long long)S * S, S,
         buf_v, (long long)S * D, HD, HD, 0, D,
         nullptr, buf_o, (long long)S * D, HD, D,
         S, HD, S, B * H, H, 1.0f);
    gemm(buf_o, 0, 0, D, WO, 0, 0, D, 0, D, bO, outp, 0, 0, D,
         (int)ROWS, D, D, 1, 1, 1.0f);
  };

  // ---------------- Encoder ----------------
  embed_kernel<<<(int)ROWS, 256, 0, stream>>>(src, emb_src, pos, buf_x, S, D);
  const float* x = buf_x;
  for (int l = 0; l < LE; ++l) {
    mha(x, D, eWQ + l * WQKV_E, eBQ + l * D, eWK + l * WQKV_E, eBK + l * D,
        eWV + l * WQKV_E, eBV + l * D, eWO + l * WSQ, eBO + l * D, 0, buf_mha);
    layernorm_kernel<<<(int)ROWS, 256, 0, stream>>>(buf_mha, x, eG + l * D,
                                                    eB + l * D, nullptr, buf_t, D);
    gemm(buf_t, 0, 0, D, eWff + l * WSQ, 0, 0, D, 0, D, eBff + l * D,
         buf_ff, 0, 0, D, (int)ROWS, D, D, 1, 1, 1.0f);
    float* eo = buf_enc + (long long)l * ROWS * D;
    layernorm_kernel<<<(int)ROWS, 256, 0, stream>>>(buf_ff, buf_t, eG + l * D,
                                                    eB + l * D, src, eo, D);
    x = eo;
  }

  // ---------------- Decoder (fed `source`, reference bug preserved) --------
  embed_kernel<<<(int)ROWS, 256, 0, stream>>>(src, emb_tgt, pos, buf_x, S, D);
  for (int l = 0; l < LD_; ++l) {
    mha(buf_x, D, mWQ + l * WQKV_E, mBQ + l * D, mWK + l * WQKV_E, mBK + l * D,
        mWV + l * WQKV_E, mBV + l * D, mWO + l * WSQ, mBO + l * D, 1, buf_mha);
    layernorm_kernel<<<(int)ROWS, 256, 0, stream>>>(buf_mha, buf_x, dG + l * D,
                                                    dB + l * D, nullptr, buf_t, D);
    concat_kernel<<<(int)ROWS, 256, 0, stream>>>(buf_enc + (long long)l * ROWS * D,
                                                 buf_t, buf_cat, D);
    mha(buf_cat, 2 * D, xWQ + l * WQKV_X, xBQ + l * D, xWK + l * WQKV_X,
        xBK + l * D, xWV + l * WQKV_X, xBV + l * D, xWO + l * WSQ, xBO + l * D,
        0, buf_mha);
    layernorm_kernel<<<(int)ROWS, 256, 0, stream>>>(buf_mha, buf_t, dG + l * D,
                                                    dB + l * D, nullptr, buf_t2, D);
    gemm(buf_t2, 0, 0, D, dWff + l * WSQ, 0, 0, D, 0, D, dBff + l * D,
         buf_ff, 0, 0, D, (int)ROWS, D, D, 1, 1, 1.0f);
    layernorm_kernel<<<(int)ROWS, 256, 0, stream>>>(buf_ff, buf_t2, dG + l * D,
                                                    dB + l * D, nullptr, buf_x, D);
  }

  // ---------------- Vocab projection: [2048,512] @ [512,32000] -------------
  gemm(buf_x, 0, 0, D, Wout, 0, 0, VT, 0, VT, bout,
       (float*)d_out, 0, 0, VT, (int)ROWS, VT, D, 1, 1, 1.0f);
}
